// MultiheadAttention_3272765080432
// MI455X (gfx1250) — matrix operand order repaired and checked by reading, not compiled, not run on hardware
//
#include <hip/hip_runtime.h>

typedef __attribute__((ext_vector_type(16))) _Float16 v16h;
typedef __attribute__((ext_vector_type(8)))  float    v8f;

#define WMMA(a, b, c) __builtin_amdgcn_wmma_f32_16x16x32_f16(false, (a), false, (b), (short)0, (c), false, false)

namespace {
constexpr int kD    = 768;
constexpr int kH    = 12;
constexpr int kDh   = 64;
constexpr int kS    = 4096;
constexpr int kB    = 2;
constexpr int kRows = kB * kS;     // 8192
constexpr int kKT   = kD / 32;     // 24 k-tiles of 32
constexpr int kNT   = kD / 16;     // 48 n-tiles of 16
}

// ---------------------------------------------------------------------------
// Fragment loaders (CDNA5 wave32 WMMA 16x16x32 f16 layouts, ISA 7.12.2)
// ---------------------------------------------------------------------------

// A-fragment 16x32 (MxK): lane L holds row (L%16); element e -> k = k0 + (L/16)*16 + e.
// One contiguous 32-byte run per lane.
__device__ __forceinline__ v16h load_a_frag(const _Float16* base, int row0, int k0,
                                            int ld, int lane) {
  const _Float16* p = base + (size_t)(row0 + (lane & 15)) * ld + (k0 + ((lane >> 4) << 4));
  union { v16h v; struct { int4 lo; int4 hi; } q; } u;
  u.q.lo = *reinterpret_cast<const int4*>(p);
  u.q.hi = *reinterpret_cast<const int4*>(p + 8);
  return u.v;
}

// B-fragment 32x16 (KxN) from a source where the contraction index k is contiguous
// in memory: src(k, n) = base[(n0 + L%16)*ld + k0 + k]; lane L holds column (L%16),
// element e -> k = (L/16)*16 + e  (one contiguous 32-byte run per lane).
__device__ __forceinline__ v16h load_b_frag_kc(const _Float16* base, int n0, int k0,
                                               int ld, int lane) {
  const _Float16* p = base + (size_t)(n0 + (lane & 15)) * ld + (k0 + ((lane >> 4) << 4));
  union { v16h v; struct { int4 lo; int4 hi; } q; } u;
  u.q.lo = *reinterpret_cast<const int4*>(p);
  u.q.hi = *reinterpret_cast<const int4*>(p + 8);
  return u.v;
}

// ---------------------------------------------------------------------------
// Kernel 1: fp32 -> f16 cast of x
// ---------------------------------------------------------------------------
__global__ void cast_f32_to_f16(const float* __restrict__ in,
                                _Float16* __restrict__ out, int n) {
  int i = blockIdx.x * blockDim.x + threadIdx.x;
  if (i < n) out[i] = (_Float16)in[i];
}

// ---------------------------------------------------------------------------
// Kernel 2: pack a [768 x 768] row-major fp32 weight into B-fragment order.
// ---------------------------------------------------------------------------
__global__ void pack_weight(const float* __restrict__ W, _Float16* __restrict__ pw) {
  int t = blockIdx.x * blockDim.x + threadIdx.x;   // kKT*kNT*32 threads
  int lane = t & 31;
  int tile = t >> 5;
  int nt = tile % kNT;
  int kt = tile / kNT;
  int n = nt * 16 + (lane & 15);
  int kbase = kt * 32 + ((lane >> 4) << 4);
  union { v16h v; _Float16 e[16]; } u;
#pragma unroll
  for (int e = 0; e < 16; ++e)
    u.e[e] = (_Float16)W[(size_t)(kbase + e) * kD + n];
  *reinterpret_cast<v16h*>(pw + (size_t)t * 16) = u.v;
}

// ---------------------------------------------------------------------------
// GEMM core: 32x64 tile per wave, ping-pong double buffer (no register
// rotation -> no v_dual_mov churn, no hazard NOP chains). kt is a full-unroll
// constant so all load addresses fold to base + immediate offset.
// ---------------------------------------------------------------------------
__device__ __forceinline__ void gemm_load_ab(const _Float16* __restrict__ aBase0,
                                             const _Float16* __restrict__ aBase1,
                                             const _Float16* __restrict__ bBase,
                                             int kt, v16h a[2], v16h b[4]) {
  union { v16h v; struct { int4 lo; int4 hi; } q; } u;
  u.q.lo = *reinterpret_cast<const int4*>(aBase0 + kt * 32);
  u.q.hi = *reinterpret_cast<const int4*>(aBase0 + kt * 32 + 8);
  a[0] = u.v;
  u.q.lo = *reinterpret_cast<const int4*>(aBase1 + kt * 32);
  u.q.hi = *reinterpret_cast<const int4*>(aBase1 + kt * 32 + 8);
  a[1] = u.v;
#pragma unroll
  for (int j = 0; j < 4; ++j)
    b[j] = *reinterpret_cast<const v16h*>(bBase + (size_t)kt * (kNT * 32 * 16) + j * 512);
}

__device__ __forceinline__ void gemm_mma8(v8f acc[8], const v16h a[2], const v16h b[4]) {
#pragma unroll
  for (int i = 0; i < 2; ++i)
#pragma unroll
    for (int j = 0; j < 4; ++j)
      acc[i * 4 + j] = WMMA(a[i], b[j], acc[i * 4 + j]);
}

__device__ __forceinline__ void gemm_32x64(const _Float16* __restrict__ X,
                                           const _Float16* __restrict__ pW,
                                           int row0, int nb, int lane, v8f acc[8]) {
  const _Float16* aBase0 = X + (size_t)(row0 + (lane & 15)) * kD + ((lane >> 4) << 4);
  const _Float16* aBase1 = aBase0 + (size_t)16 * kD;
  const _Float16* bBase  = pW + ((size_t)(nb * 4) * 32 + lane) * 16;

  v16h A0[2], B0[4], A1[2], B1[4];
  gemm_load_ab(aBase0, aBase1, bBase, 0, A0, B0);
#pragma unroll
  for (int kt = 0; kt < kKT - 2; kt += 2) {
    gemm_load_ab(aBase0, aBase1, bBase, kt + 1, A1, B1);  // loads overlap...
    gemm_mma8(acc, A0, B0);                               // ...these WMMAs
    gemm_load_ab(aBase0, aBase1, bBase, kt + 2, A0, B0);
    gemm_mma8(acc, A1, B1);
  }
  gemm_load_ab(aBase0, aBase1, bBase, kKT - 1, A1, B1);
  gemm_mma8(acc, A0, B0);
  gemm_mma8(acc, A1, B1);
}

// ---------------------------------------------------------------------------
// Kernel 3: QKV projection. grid (32, 12 heads, 3 matrices), block 256 (8 waves).
// Q gets 1/sqrt(Dh) folded in; V is stored transposed [B,H,Dh,S].
// ---------------------------------------------------------------------------
__global__ void __launch_bounds__(256)
qkv_gemm(const _Float16* __restrict__ Xh,
         const _Float16* __restrict__ pWq, const _Float16* __restrict__ pWk,
         const _Float16* __restrict__ pWv,
         _Float16* __restrict__ Qh, _Float16* __restrict__ Kh,
         _Float16* __restrict__ Vth) {
  int wave = threadIdx.x >> 5;
  int lane = threadIdx.x & 31;
  int row0 = (blockIdx.x * 8 + wave) * 32;   // 32-row tile, 0..8191
  int nb = blockIdx.y;                       // head 0..11
  int z  = blockIdx.z;                       // 0=Q 1=K 2=V
  const _Float16* pW = (z == 0) ? pWq : (z == 1) ? pWk : pWv;

  v8f acc[8] = {};
  gemm_32x64(Xh, pW, row0, nb, lane, acc);

  int mhalf = (lane >> 4) << 3;
#pragma unroll
  for (int i = 0; i < 2; ++i) {
#pragma unroll
    for (int r = 0; r < 8; ++r) {
      int m = row0 + i * 16 + r + mhalf;     // global row in [0, 8192)
      int bb = m >> 12;
      int s = m & (kS - 1);
#pragma unroll
      for (int nt2 = 0; nt2 < 4; ++nt2) {
        int dh = nt2 * 16 + (lane & 15);
        float v = acc[i * 4 + nt2][r];
        if (z == 0) {
          Qh[(((size_t)(bb * kH + nb)) * kS + s) * kDh + dh] = (_Float16)(v * 0.125f);
        } else if (z == 1) {
          Kh[(((size_t)(bb * kH + nb)) * kS + s) * kDh + dh] = (_Float16)v;
        } else {
          Vth[(((size_t)(bb * kH + nb)) * kDh + dh) * kS + s] = (_Float16)v;
        }
      }
    }
  }
}

// ---------------------------------------------------------------------------
// Kernel 4: causal flash attention. grid (S/64, H, B), block 128 (4 waves).
// The workgroup owns 64 query rows (one 16-row tile per wave). Each 32-key
// block of K (32x64) and V^T (64x32) is staged into LDS ONCE by all 128
// threads, then all 4 waves build WMMA fragments from LDS -> 4x less L2
// traffic. Online softmax keeps scores/O in VGPRs; P crosses C-layout ->
// A-layout through a per-wave LDS buffer with a wave-local s_wait_dscnt.
// ---------------------------------------------------------------------------
__global__ void __launch_bounds__(128)
attn_flash(const _Float16* __restrict__ Qh, const _Float16* __restrict__ Kh,
           const _Float16* __restrict__ Vth, _Float16* __restrict__ Ctxh) {
  int tid  = threadIdx.x;
  int wave = tid >> 5;
  int lane = tid & 31;
  int m0wg = blockIdx.x * 64;
  int m0   = m0wg + wave * 16;
  int h    = blockIdx.y;
  int b    = blockIdx.z;
  const _Float16* Q  = Qh  + ((size_t)(b * kH + h)) * kS * kDh;
  const _Float16* K  = Kh  + ((size_t)(b * kH + h)) * kS * kDh;
  const _Float16* Vt = Vth + ((size_t)(b * kH + h)) * kDh * kS;

  __shared__ __align__(16) _Float16 ldsK[32 * 64];          // [key_local][dh]
  __shared__ __align__(16) _Float16 ldsV[64 * 32];          // [dh][key_local]
  __shared__ __align__(16) _Float16 ldsP[4][16 * 32];       // per-wave P tile

  v16h qa0 = load_a_frag(Q, m0, 0, kDh, lane);
  v16h qa1 = load_a_frag(Q, m0, 32, kDh, lane);

  float mst[8], lst[8];
  v8f o0 = {}, o1 = {}, o2 = {}, o3 = {};
#pragma unroll
  for (int r = 0; r < 8; ++r) { mst[r] = -1e30f; lst[r] = 0.0f; }

  int mhalf = (lane >> 4) << 3;
  // copy-role indices (same every iteration)
  int krow = tid >> 2, kch = tid & 3;        // K tile: 32 rows x 128B, 4 thr/row
  int vrow = tid >> 1, vch = tid & 1;        // V tile: 64 rows x 64B,  2 thr/row

  int nblk = (m0wg >> 5) + 2;                // key blocks of 32 covering m0wg+63
  for (int jb = 0; jb < nblk; ++jb) {
    int j0 = jb * 32;
    __syncthreads();                         // previous block fully consumed
    {
      const int4* srcK = reinterpret_cast<const int4*>(K + (size_t)(j0 + krow) * kDh);
      int4* dstK = reinterpret_cast<int4*>(ldsK + krow * 64);
      dstK[kch * 2]     = srcK[kch * 2];
      dstK[kch * 2 + 1] = srcK[kch * 2 + 1];
      const int4* srcV = reinterpret_cast<const int4*>(Vt + (size_t)vrow * kS + j0);
      int4* dstV = reinterpret_cast<int4*>(ldsV + vrow * 32);
      dstV[vch * 2]     = srcV[vch * 2];
      dstV[vch * 2 + 1] = srcV[vch * 2 + 1];
      if (jb + 1 < nblk) {                   // prefetch next block's lines into L2
        __builtin_prefetch(K + (size_t)(j0 + 32 + krow) * kDh + kch * 16, 0, 1);
        __builtin_prefetch(Vt + (size_t)vrow * kS + j0 + 32, 0, 1);
      }
    }
    __syncthreads();                         // tiles visible to all waves

    if (j0 <= m0 + 15) {                     // causal: this wave still has keys here
      // ---- scores: S = Q (16x64) @ K^T (64x32)
      v16h kb00 = load_b_frag_kc(ldsK, 0,  0,  kDh, lane);
      v16h kb01 = load_b_frag_kc(ldsK, 0,  32, kDh, lane);
      v16h kb10 = load_b_frag_kc(ldsK, 16, 0,  kDh, lane);
      v16h kb11 = load_b_frag_kc(ldsK, 16, 32, kDh, lane);
      v8f s0 = {}, s1 = {};
      s0 = WMMA(qa0, kb00, s0);
      s0 = WMMA(qa1, kb01, s0);
      s1 = WMMA(qa0, kb10, s1);
      s1 = WMMA(qa1, kb11, s1);

      // ---- causal mask (only near the diagonal)
      if (j0 + 31 >= m0) {
        int kcol = j0 + (lane & 15);
#pragma unroll
        for (int r = 0; r < 8; ++r) {
          int q = m0 + r + mhalf;
          if (kcol > q)      s0[r] = -1e30f;
          if (kcol + 16 > q) s1[r] = -1e30f;
        }
      }

      // ---- online softmax: row max/sum via shuffles within 16-lane halves
      _Float16* Pw = &ldsP[wave][0];
#pragma unroll
      for (int r = 0; r < 8; ++r) {
        float vmax = fmaxf(s0[r], s1[r]);
        vmax = fmaxf(vmax, __shfl_xor(vmax, 1));
        vmax = fmaxf(vmax, __shfl_xor(vmax, 2));
        vmax = fmaxf(vmax, __shfl_xor(vmax, 4));
        vmax = fmaxf(vmax, __shfl_xor(vmax, 8));
        float mnew  = fmaxf(mst[r], vmax);
        float alpha = __expf(mst[r] - mnew);
        float p0 = __expf(s0[r] - mnew);
        float p1 = __expf(s1[r] - mnew);
        float rs = p0 + p1;
        rs += __shfl_xor(rs, 1);
        rs += __shfl_xor(rs, 2);
        rs += __shfl_xor(rs, 4);
        rs += __shfl_xor(rs, 8);
        lst[r] = lst[r] * alpha + rs;
        mst[r] = mnew;
        o0[r] *= alpha; o1[r] *= alpha; o2[r] *= alpha; o3[r] *= alpha;
        int mrow = r + mhalf;
        Pw[mrow * 32 + (lane & 15)]      = (_Float16)p0;
        Pw[mrow * 32 + (lane & 15) + 16] = (_Float16)p1;
      }
      // wave-local DS ordering: P stores complete before the A-fragment reload
      asm volatile("s_wait_dscnt 0x0" ::: "memory");

      // ---- O += P (16x32) @ V (32x64)
      v16h pa  = load_a_frag(Pw, 0, 0, 32, lane);
      v16h vb0 = load_b_frag_kc(ldsV, 0,  0, 32, lane);
      v16h vb1 = load_b_frag_kc(ldsV, 16, 0, 32, lane);
      v16h vb2 = load_b_frag_kc(ldsV, 32, 0, 32, lane);
      v16h vb3 = load_b_frag_kc(ldsV, 48, 0, 32, lane);
      o0 = WMMA(pa, vb0, o0);
      o1 = WMMA(pa, vb1, o1);
      o2 = WMMA(pa, vb2, o2);
      o3 = WMMA(pa, vb3, o3);
    }
  }

  // ---- normalize and write ctx in [B, S, H*Dh] layout for the output GEMM
#pragma unroll
  for (int r = 0; r < 8; ++r) {
    int s = m0 + r + mhalf;
    float invl = 1.0f / lst[r];
    size_t off = ((size_t)b * kS + s) * kD + h * kDh + (lane & 15);
    Ctxh[off]      = (_Float16)(o0[r] * invl);
    Ctxh[off + 16] = (_Float16)(o1[r] * invl);
    Ctxh[off + 32] = (_Float16)(o2[r] * invl);
    Ctxh[off + 48] = (_Float16)(o3[r] * invl);
  }
}

// ---------------------------------------------------------------------------
// Kernel 5: output projection ctx @ Wo + bo -> fp32 out. 32x64/wave, pipelined.
// ---------------------------------------------------------------------------
__global__ void __launch_bounds__(256)
out_gemm(const _Float16* __restrict__ Ctxh, const _Float16* __restrict__ pWo,
         const float* __restrict__ bo, float* __restrict__ out) {
  int wave = threadIdx.x >> 5;
  int lane = threadIdx.x & 31;
  int row0 = (blockIdx.x * 8 + wave) * 32;
  int nb = blockIdx.y;

  v8f acc[8] = {};
  gemm_32x64(Ctxh, pWo, row0, nb, lane, acc);

  int mhalf = (lane >> 4) << 3;
  int ncol = nb * 64 + (lane & 15);
  float bias0 = bo[ncol], bias1 = bo[ncol + 16], bias2 = bo[ncol + 32], bias3 = bo[ncol + 48];
#pragma unroll
  for (int i = 0; i < 2; ++i) {
#pragma unroll
    for (int r = 0; r < 8; ++r) {
      int m = row0 + i * 16 + r + mhalf;
      size_t off = (size_t)m * kD + ncol;
      out[off]      = acc[i * 4 + 0][r] + bias0;
      out[off + 16] = acc[i * 4 + 1][r] + bias1;
      out[off + 32] = acc[i * 4 + 2][r] + bias2;
      out[off + 48] = acc[i * 4 + 3][r] + bias3;
    }
  }
}

// ---------------------------------------------------------------------------
extern "C" void kernel_launch(void* const* d_in, const int* in_sizes, int n_in,
                              void* d_out, int out_size, void* d_ws, size_t ws_size,
                              hipStream_t stream) {
  (void)in_sizes; (void)n_in; (void)out_size; (void)ws_size;
  const float* x  = (const float*)d_in[0];
  const float* Wq = (const float*)d_in[1];
  const float* Wk = (const float*)d_in[2];
  const float* Wv = (const float*)d_in[3];
  const float* Wo = (const float*)d_in[4];
  const float* bo = (const float*)d_in[5];
  float* out = (float*)d_out;

  const size_t XH_BYTES = (size_t)kRows * kD * sizeof(_Float16);          // 12.6 MB
  const size_t PW_BYTES = (size_t)kKT * kNT * 32 * 16 * sizeof(_Float16); // 1.125 MB
  char* ws = (char*)d_ws;
  _Float16* Xh   = (_Float16*)ws;              ws += XH_BYTES;
  _Float16* pWq  = (_Float16*)ws;              ws += PW_BYTES;
  _Float16* pWk  = (_Float16*)ws;              ws += PW_BYTES;
  _Float16* pWv  = (_Float16*)ws;              ws += PW_BYTES;
  _Float16* pWo  = (_Float16*)ws;              ws += PW_BYTES;
  _Float16* Qh   = (_Float16*)ws;              ws += XH_BYTES;
  _Float16* Kh   = (_Float16*)ws;              ws += XH_BYTES;
  _Float16* Vth  = (_Float16*)ws;              ws += XH_BYTES;
  _Float16* Ctxh = (_Float16*)ws;              ws += XH_BYTES;

  const int nElem = kRows * kD;                            // 6,291,456
  cast_f32_to_f16<<<(nElem + 255) / 256, 256, 0, stream>>>(x, Xh, nElem);

  const int packThreads = kKT * kNT * 32;                  // 36,864
  pack_weight<<<packThreads / 256, 256, 0, stream>>>(Wq, pWq);
  pack_weight<<<packThreads / 256, 256, 0, stream>>>(Wk, pWk);
  pack_weight<<<packThreads / 256, 256, 0, stream>>>(Wv, pWv);
  pack_weight<<<packThreads / 256, 256, 0, stream>>>(Wo, pWo);

  qkv_gemm<<<dim3(kRows / 32 / 8, kH, 3), 256, 0, stream>>>(Xh, pWq, pWk, pWv,
                                                            Qh, Kh, Vth);

  attn_flash<<<dim3(kS / 64, kH, kB), 128, 0, stream>>>(Qh, Kh, Vth, Ctxh);

  out_gemm<<<dim3(kRows / 32 / 8, kH, 1), 256, 0, stream>>>(Ctxh, pWo, bo, out);
}
